// Interpolation_62096637166373
// MI455X (gfx1250) — matrix operand order repaired and checked by reading, not compiled, not run on hardware
//
#include <hip/hip_runtime.h>
#include <hip/hip_bf16.h>

// ---------------------------------------------------------------------------
// Problem constants (from the reference)
// ---------------------------------------------------------------------------
#define BATCH 4
#define NPTS  2048
#define MSEED 512
#define KINT  16
#define KSUB  8
#define NUP   (NPTS * 2)

static constexpr float kBnScale = 0.9999950000374997f; // 1/sqrt(1+1e-5)

// ---------------------------------------------------------------------------
// WMMA types (CDNA5 / gfx1250, wave32)
// ---------------------------------------------------------------------------
typedef __attribute__((ext_vector_type(16))) __bf16 bf16x16;
typedef __attribute__((ext_vector_type(8)))  __bf16 bf16x8;
typedef __attribute__((ext_vector_type(2)))  __bf16 bf16x2;
typedef __attribute__((ext_vector_type(8)))  float  f32x8;

// ---------------------------------------------------------------------------
// Fused WMMA GEMM:  Y[b] = epi( W(MxK) @ X[b](K x Ncols) )
//   epi: +bias[m]  -> *(g[m]*kBnScale)+be[m]  -> relu   (each optional)
//
// LDS layouts are arranged so each lane's WMMA fragment is CONTIGUOUS:
//  A: [row][k] (lanes<16 hold K0..15, lanes>=16 K16..31), same order as B
//  B: [col][k] transposed-in-LDS (lanes<16 hold K0..15, lanes>=16 K16..31)
// -> fragment build is 2x ds_load_b128 per matrix instead of 16x ds_load_u16.
// Row stride 40 bf16 (20 dwords) keeps 16B alignment and spreads banks.
//
// Block: 256 thr = 8 waves; block tile 32(M) x 128(N); wave tile 16x32
// (A-fragment reused across 2 accumulators -> 2 v_wmma per K-step per wave).
// Every launch in this pipeline has Ncols % 128 == 0; big layers have K%32==0
// so the unguarded fast staging path (packed bf16x2 stores) dominates.
// Edge tiles use CLAMPED-ADDRESS branch-free staging (load from a safe
// address, select 0) to avoid per-element exec-mask serialization.
// ---------------------------------------------------------------------------
#define TM 32
#define TN 128
#define TK 32
#define LDA 40
#define LDB 40
#define ASLOT(k) (((k) & 7) | (((k) & 8) << 1) | (((k) & 16) >> 1))

__global__ __launch_bounds__(256)
void wmma_gemm_kernel(const float* __restrict__ W, const float* __restrict__ X,
                      float* __restrict__ Y, const float* __restrict__ bias,
                      const float* __restrict__ bn_g, const float* __restrict__ bn_b,
                      int M, int K, int Ncols, int relu)
{
  __shared__ __bf16 As[TM * LDA];
  __shared__ __bf16 Bs[TN * LDB];

  const int b    = blockIdx.z;
  const int m0   = blockIdx.x * TM;
  const int n0   = blockIdx.y * TN;
  const int tid  = threadIdx.x;
  const int lane = tid & 31;          // wave32
  const int wave = tid >> 5;
  const int tm   = (wave & 1) * 16;
  const int tn   = (wave >> 1) * 32;

  const float* Xb = X + (size_t)b * K * Ncols;
  float*       Yb = Y + (size_t)b * M * Ncols;

  f32x8 acc0 = {}, acc1 = {};
  const bool mfull = (m0 + TM <= M);

  for (int k0 = 0; k0 < K; k0 += TK) {
    const bool kfull = (k0 + TK <= K);

    // ---- stage A tile (32 x 32) ----
    if (mfull && kfull) {
#pragma unroll
      for (int i = 0; i < 2; ++i) {
        int p = tid + i * 256;                 // pair id 0..511
        int r = p >> 4, k = (p & 15) * 2;
        const float* src = &W[(size_t)(m0 + r) * K + (k0 + k)];
        float v0 = src[0], v1 = src[1];        // -> global_load_b64
        bf16x2 pk; pk[0] = (__bf16)v0; pk[1] = (__bf16)v1;
        *(bf16x2*)&As[r * LDA + k] = pk; // -> cvt_pk + ds_store_b32
      }
    } else {
      // branch-free edge path: clamp address to a valid element, select 0
#pragma unroll
      for (int i = 0; i < 4; ++i) {
        int e = tid + i * 256;
        int r = e >> 5, c = e & 31;
        int gm = m0 + r, gk = k0 + c;
        bool ok = (gm < M) & (gk < K);
        size_t off = ok ? ((size_t)gm * K + gk) : 0;
        float v = W[off];
        As[r * LDA + c] = (__bf16)(ok ? v : 0.0f);
      }
    }

    // ---- stage B tile (32 x 128), transposed into [col][k] ----
    if (kfull) {
#pragma unroll
      for (int i = 0; i < 8; ++i) {
        int q = tid + i * 256;                 // pair id 0..2047
        int col = q & (TN - 1), k = (q >> 7) * 2;
        const float* src = &Xb[(size_t)(k0 + k) * Ncols + (n0 + col)];
        float v0 = src[0];
        float v1 = src[Ncols];
        bf16x2 pk; pk[0] = (__bf16)v0; pk[1] = (__bf16)v1;
        *(bf16x2*)&Bs[col * LDB + k] = pk;
      }
    } else {
      // branch-free edge path (n0+col < Ncols always: Ncols % TN == 0)
#pragma unroll
      for (int i = 0; i < 16; ++i) {
        int e = tid + i * 256;                 // elem id 0..4095
        int col = e & (TN - 1), kk = e >> 7;
        int gk = k0 + kk;
        bool ok = (gk < K);
        float v = Xb[(size_t)(ok ? gk : 0) * Ncols + (n0 + col)];
        Bs[col * LDB + kk] = (__bf16)(ok ? v : 0.0f);
      }
    }

    // prefetch next K tile of X (global_prefetch_b8)
    if (k0 + TK < K) {
      int gn = n0 + (tid & 127);
      __builtin_prefetch(&Xb[(size_t)(k0 + TK) * Ncols + gn], 0, 1);
    }
    __syncthreads();

    // ---- build fragments: 2x ds_load_b128 each ----
    const int grp = lane >> 4;
    const int mr  = tm + (lane & 15);
    const __bf16* pa = &As[mr * LDA + grp * 16];
    bf16x8 alo = *(const bf16x8*)pa;
    bf16x8 ahi = *(const bf16x8*)(pa + 8);
    bf16x16 af = __builtin_shufflevector(alo, ahi,
        0, 1, 2, 3, 4, 5, 6, 7, 8, 9, 10, 11, 12, 13, 14, 15);

    const int nc = tn + (lane & 15);
    const __bf16* pb0 = &Bs[nc * LDB + grp * 16];
    bf16x8 b0lo = *(const bf16x8*)pb0;
    bf16x8 b0hi = *(const bf16x8*)(pb0 + 8);
    bf16x16 bf0 = __builtin_shufflevector(b0lo, b0hi,
        0, 1, 2, 3, 4, 5, 6, 7, 8, 9, 10, 11, 12, 13, 14, 15);
    const __bf16* pb1 = pb0 + 16 * LDB;
    bf16x8 b1lo = *(const bf16x8*)pb1;
    bf16x8 b1hi = *(const bf16x8*)(pb1 + 8);
    bf16x16 bf1 = __builtin_shufflevector(b1lo, b1hi,
        0, 1, 2, 3, 4, 5, 6, 7, 8, 9, 10, 11, 12, 13, 14, 15);

    acc0 = __builtin_amdgcn_wmma_f32_16x16x32_bf16(false, af, false, bf0,
                                                   (short)0, acc0, false, false);
    acc1 = __builtin_amdgcn_wmma_f32_16x16x32_bf16(false, af, false, bf1,
                                                   (short)0, acc1, false, false);
    __syncthreads();
  }

  // ---- epilogue (C/D 16x16 f32: VGPR j -> M = j + 8*(lane>=16), N = lane&15) ----
  const int nc0 = n0 + tn + (lane & 15);
  const int nc1 = nc0 + 16;
#pragma unroll
  for (int j = 0; j < 8; ++j) {
    int mr = m0 + tm + (lane >> 4) * 8 + j;
    if (mr < M) {
      float bi = bias ? bias[mr] : 0.0f;
      float sg = bn_g ? (bn_g[mr] * kBnScale) : 1.0f;
      float sb = bn_g ? bn_b[mr] : 0.0f;
      float y0 = (acc0[j] + bi) * sg + sb;
      float y1 = (acc1[j] + bi) * sg + sb;
      if (relu) { y0 = fmaxf(y0, 0.0f); y1 = fmaxf(y1, 0.0f); }
      if (nc0 < Ncols) Yb[(size_t)mr * Ncols + nc0] = y0;
      if (nc1 < Ncols) Yb[(size_t)mr * Ncols + nc1] = y1;
    }
  }
}

// ---------------------------------------------------------------------------
// Layout helpers
// ---------------------------------------------------------------------------
__global__ void pos_to_points_kernel(const float* __restrict__ pos,
                                     float* __restrict__ pts, int N)
{ // (B,3,N) -> (B,N,3)
  int b = blockIdx.y, n = blockIdx.x * 256 + threadIdx.x;
  if (n >= N) return;
  for (int j = 0; j < 3; ++j)
    pts[((size_t)b * N + n) * 3 + j] = pos[((size_t)b * 3 + j) * N + n];
}

__global__ void points_to_ch_kernel(const float* __restrict__ pts,
                                    float* __restrict__ out, int N)
{ // (B,N,3) -> (B,3,N)
  int b = blockIdx.y, n = blockIdx.x * 256 + threadIdx.x;
  if (n >= N) return;
  for (int j = 0; j < 3; ++j)
    out[((size_t)b * 3 + j) * N + n] = pts[((size_t)b * N + n) * 3 + j];
}

// ---------------------------------------------------------------------------
// KNN (brute force, register insertion sort; stable for ties like jax top_k)
// ---------------------------------------------------------------------------
template <int KNN>
__global__ void knn_kernel(const float* __restrict__ query, // (B,Nq,3)
                           const float* __restrict__ ref,   // (B,Nr,3)
                           int* __restrict__ idx, int Nq, int Nr)
{
  int b = blockIdx.y, t = blockIdx.x * 256 + threadIdx.x;
  if (t >= Nq) return;
  const float* q = query + ((size_t)b * Nq + t) * 3;
  float qx = q[0], qy = q[1], qz = q[2];
  float dbest[KNN]; int ibest[KNN];
#pragma unroll
  for (int i = 0; i < KNN; ++i) { dbest[i] = 3.4e38f; ibest[i] = 0; }
  const float* rb = ref + (size_t)b * Nr * 3;
  for (int r = 0; r < Nr; ++r) {
    float dx = qx - rb[3 * r], dy = qy - rb[3 * r + 1], dz = qz - rb[3 * r + 2];
    float dn = dx * dx + dy * dy + dz * dz;
    int in = r;
#pragma unroll
    for (int i = 0; i < KNN; ++i) {
      if (dn < dbest[i]) {
        float td = dbest[i]; int ti = ibest[i];
        dbest[i] = dn; ibest[i] = in; dn = td; in = ti;
      }
    }
  }
  int* o = idx + ((size_t)b * Nq + t) * KNN;
#pragma unroll
  for (int i = 0; i < KNN; ++i) o[i] = ibest[i];
}

// ---------------------------------------------------------------------------
// Grouping (gather) with fused broadcast subtraction.
// mode 0: out = g ; mode 1: out = bcast - g ; mode 2: out = g - bcast
// ---------------------------------------------------------------------------
__global__ void group_kernel(const float* __restrict__ fea, const int* __restrict__ idx,
                             const float* __restrict__ bcast, float* __restrict__ out,
                             int C, int Nsrc, int Nq, int K, int mode)
{
  int b = blockIdx.y, t = blockIdx.x * 256 + threadIdx.x;
  if (t >= Nq * K) return;
  int n = t / K, kk = t - n * K;
  int src = idx[((size_t)b * Nq + n) * K + kk];
  const float* fb = fea + (size_t)b * C * Nsrc;
  const float* bb = bcast ? bcast + (size_t)b * C * Nq : nullptr;
  float* ob = out + (size_t)b * C * Nq * K;
  for (int c = 0; c < C; ++c) {
    float g = fb[(size_t)c * Nsrc + src];
    float r = g;
    if (mode == 1) r = bb[(size_t)c * Nq + n] - g;
    else if (mode == 2) r = g - bb[(size_t)c * Nq + n];
    ob[((size_t)c * Nq + n) * K + kk] = r;
  }
}

// ---------------------------------------------------------------------------
// Elementwise / reduction helpers
// ---------------------------------------------------------------------------
__global__ void ew_add_kernel(float* o, const float* a, const float* b, size_t n)
{ size_t t = (size_t)blockIdx.x * 256 + threadIdx.x; if (t < n) o[t] = a[t] + b[t]; }
__global__ void ew_avg_kernel(float* o, const float* a, const float* b, size_t n)
{ size_t t = (size_t)blockIdx.x * 256 + threadIdx.x; if (t < n) o[t] = 0.5f * (a[t] + b[t]); }
__global__ void ew_sub_kernel(float* o, const float* a, const float* b, size_t n)
{ size_t t = (size_t)blockIdx.x * 256 + threadIdx.x; if (t < n) o[t] = a[t] - b[t]; }
__global__ void ew_relu_kernel(float* o, const float* a, size_t n)
{ size_t t = (size_t)blockIdx.x * 256 + threadIdx.x; if (t < n) o[t] = fmaxf(a[t], 0.0f); }

__global__ void softmax_rows_kernel(float* w, size_t rows, int K)
{
  size_t t = (size_t)blockIdx.x * 256 + threadIdx.x;
  if (t >= rows) return;
  float* r = w + t * K;
  float mx = r[0];
  for (int k = 1; k < K; ++k) mx = fmaxf(mx, r[k]);
  float s = 0.0f;
  for (int k = 0; k < K; ++k) { r[k] = __expf(r[k] - mx); s += r[k]; }
  float inv = 1.0f / s;
  for (int k = 0; k < K; ++k) r[k] *= inv;
}

__global__ void attn_agg_kernel(const float* __restrict__ w, const float* __restrict__ v,
                                float* __restrict__ out, size_t rows, int K)
{
  size_t t = (size_t)blockIdx.x * 256 + threadIdx.x;
  if (t >= rows) return;
  const float* wr = w + t * K; const float* vr = v + t * K;
  float s = 0.0f;
  for (int k = 0; k < K; ++k) s += wr[k] * vr[k];
  out[t] = s;
}

__global__ void rowmax_kernel(const float* __restrict__ x, float* __restrict__ o,
                              int rows, int N)
{
  int r = blockIdx.x * 256 + threadIdx.x;
  if (r >= rows) return;
  const float* xr = x + (size_t)r * N;
  float m = xr[0];
  for (int n = 1; n < N; ++n) m = fmaxf(m, xr[n]);
  o[r] = m;
}

__global__ void copy_ch_kernel(float* __restrict__ out, int Cout, int coff,
                               const float* __restrict__ in, int Cin, int Ncols)
{
  int b = blockIdx.y;
  size_t t = (size_t)blockIdx.x * 256 + threadIdx.x;
  if (t >= (size_t)Cin * Ncols) return;
  int c = (int)(t / Ncols), n = (int)(t % Ncols);
  out[((size_t)b * Cout + coff + c) * Ncols + n] = in[((size_t)b * Cin + c) * Ncols + n];
}

__global__ void bcast_ch_kernel(float* __restrict__ out, int Cout, int coff,
                                const float* __restrict__ in, int Cin, int Ncols)
{ // in: (B,Cin) broadcast over Ncols
  int b = blockIdx.y;
  size_t t = (size_t)blockIdx.x * 256 + threadIdx.x;
  if (t >= (size_t)Cin * Ncols) return;
  int c = (int)(t / Ncols), n = (int)(t % Ncols);
  out[((size_t)b * Cout + coff + c) * Ncols + n] = in[b * Cin + c];
}

__global__ void build_h_kernel(const float* __restrict__ prel, float* __restrict__ h, int NK)
{ // prel (B,3,NK) -> h (B,4,NK) = [norm, x, y, z]
  int b = blockIdx.y, t = blockIdx.x * 256 + threadIdx.x;
  if (t >= NK) return;
  float x = prel[((size_t)b * 3 + 0) * NK + t];
  float y = prel[((size_t)b * 3 + 1) * NK + t];
  float z = prel[((size_t)b * 3 + 2) * NK + t];
  h[((size_t)b * 4 + 0) * NK + t] = sqrtf(x * x + y * y + z * z);
  h[((size_t)b * 4 + 1) * NK + t] = x;
  h[((size_t)b * 4 + 2) * NK + t] = y;
  h[((size_t)b * 4 + 3) * NK + t] = z;
}

__global__ void pack_wmT_kernel(const float* __restrict__ wmTW,
                                float* __restrict__ W0, float* __restrict__ W1)
{ // wmTW (256,64,2) -> Wr (64x256 row-major): Wr[d*256+c] = wmTW[c,d,r]
  int t = blockIdx.x * 256 + threadIdx.x;
  if (t >= 64 * 256) return;
  int d = t >> 8, c = t & 255;
  W0[t] = wmTW[((size_t)c * 64 + d) * 2 + 0];
  W1[t] = wmTW[((size_t)c * 64 + d) * 2 + 1];
}

// Fused: softmax(score_r) over k (bias drops by shift invariance) dotted with
// relative values vg; writes upsampled agg (B,64,2N).
__global__ void upsample_attn_kernel(const float* __restrict__ s0, const float* __restrict__ s1,
                                     const float* __restrict__ vg, float* __restrict__ agg,
                                     size_t rows, int Nn)
{
  size_t t = (size_t)blockIdx.x * 256 + threadIdx.x;
  if (t >= rows) return;
  size_t bd = t / Nn, n = t % Nn;
  const float* v = vg + t * 16;
  for (int r = 0; r < 2; ++r) {
    const float* s = (r ? s1 : s0) + t * 16;
    float mx = s[0];
#pragma unroll
    for (int k = 1; k < 16; ++k) mx = fmaxf(mx, s[k]);
    float e[16]; float sum = 0.0f;
#pragma unroll
    for (int k = 0; k < 16; ++k) { e[k] = __expf(s[k] - mx); sum += e[k]; }
    float inv = 1.0f / sum, dot = 0.0f;
#pragma unroll
    for (int k = 0; k < 16; ++k) dot += e[k] * inv * v[k];
    agg[bd * (size_t)(2 * Nn) + 2 * n + r] = dot;
  }
}

__global__ void repeat2_kernel(const float* __restrict__ in, float* __restrict__ out,
                               int rows, int N)
{
  size_t t = (size_t)blockIdx.x * 256 + threadIdx.x;
  if (t >= (size_t)rows * N) return;
  size_t r = t / N, n = t % N;
  float v = in[t];
  out[r * (size_t)(2 * N) + 2 * n + 0] = v;
  out[r * (size_t)(2 * N) + 2 * n + 1] = v;
}

__global__ void pcd_kernel(float* __restrict__ out, const float* __restrict__ repp,
                           const float* __restrict__ mo, size_t n)
{
  size_t t = (size_t)blockIdx.x * 256 + threadIdx.x;
  if (t < n) out[t] = repp[t] + 0.2f * tanhf(mo[t]);
}

// ---------------------------------------------------------------------------
// Host-side orchestration
// ---------------------------------------------------------------------------
static inline unsigned cdiv(size_t a, size_t b) { return (unsigned)((a + b - 1) / b); }

struct Arena {
  char* base; size_t off;
  float* f(size_t n) { size_t o = (off + 255) & ~(size_t)255; off = o + n * 4; return (float*)(base + o); }
  int*   i(size_t n) { size_t o = (off + 255) & ~(size_t)255; off = o + n * 4; return (int*)(base + o); }
};

static void gemm(hipStream_t st, const float* W, const float* X, float* Y,
                 const float* bias, const float* g, const float* be,
                 int M, int K, int Ncols, int relu)
{
  dim3 grid(cdiv(M, TM), cdiv(Ncols, TN), BATCH);
  wmma_gemm_kernel<<<grid, 256, 0, st>>>(W, X, Y, bias, g, be, M, K, Ncols, relu);
}

static void ew_add(hipStream_t st, float* o, const float* a, const float* b, size_t n)
{ ew_add_kernel<<<cdiv(n, 256), 256, 0, st>>>(o, a, b, n); }
static void ew_avg(hipStream_t st, float* o, const float* a, const float* b, size_t n)
{ ew_avg_kernel<<<cdiv(n, 256), 256, 0, st>>>(o, a, b, n); }
static void group(hipStream_t st, const float* fea, const int* idx, const float* bc,
                  float* out, int C, int Nsrc, int Nq, int K, int mode)
{ group_kernel<<<dim3(cdiv((size_t)Nq * K, 256), BATCH), 256, 0, st>>>(fea, idx, bc, out, C, Nsrc, Nq, K, mode); }
static void copy_ch(hipStream_t st, float* out, int Cout, int coff, const float* in, int Cin, int Ncols)
{ copy_ch_kernel<<<dim3(cdiv((size_t)Cin * Ncols, 256), BATCH), 256, 0, st>>>(out, Cout, coff, in, Cin, Ncols); }

extern "C" void kernel_launch(void* const* d_in, const int* in_sizes, int n_in,
                              void* d_out, int out_size, void* d_ws, size_t ws_size,
                              hipStream_t stream)
{
  (void)in_sizes; (void)n_in; (void)out_size; (void)ws_size;
  const float* pos       = (const float*)d_in[0]; // (B,3,N)
  const float* fea       = (const float*)d_in[1]; // (B,128,N)
  const float* seed      = (const float*)d_in[2]; // (B,512,3)
  const float* seed_feat = (const float*)d_in[3]; // (B,128,512)
#define PP(i) ((const float*)d_in[4 + (i)])
  const float *ca_vm1W=PP(0),*ca_vm1b=PP(1),*ca_vm2W=PP(2),*ca_vm2b=PP(3),
    *ca_qW=PP(4),*ca_kW=PP(5),*ca_vW=PP(6),*ca_vb=PP(7),
    *ca_pm1W=PP(8),*ca_pm1b=PP(9),*ca_pm_g=PP(10),*ca_pm_be=PP(11),
    *ca_pm2W=PP(12),*ca_pm2b=PP(13),
    *ca_am1W=PP(14),*ca_am1b=PP(15),*ca_am_g=PP(16),*ca_am_be=PP(17),
    *ca_am2W=PP(18),*ca_am2b=PP(19),*ca_eW=PP(20),*ca_eb=PP(21),
    *pr_sW=PP(22),*pr_sb=PP(23),*pr_kW=PP(24),*pr_kb=PP(25),
    *pr_vW=PP(26),*pr_vb=PP(27),*pr_qW=PP(28),*pr_qb=PP(29),
    *pr_pm1W=PP(30),*pr_pm1b=PP(31),*pr_pm_g=PP(32),*pr_pm_be=PP(33),
    *pr_pm2W=PP(34),*pr_pm2b=PP(35),
    *pr_am1W=PP(36),*pr_am1b=PP(37),*pr_am_g=PP(38),*pr_am_be=PP(39),
    *pr_am2W=PP(40),*pr_am2b=PP(41),*pr_eW=PP(42),*pr_eb=PP(43),
    *upW=PP(44),*upb=PP(45),
    *m1aW=PP(46),*m1ab=PP(47),*m1bW=PP(48),*m1bb=PP(49),
    *m2aW=PP(50),*m2ab=PP(51),*m2bW=PP(52),*m2bb=PP(53),
    *map_1W=PP(54),*map_1b=PP(55),*map_2W=PP(56),*map_2b=PP(57),
    *map_sW=PP(58),*map_sb=PP(59),
    *rev_1W=PP(60),*rev_1b=PP(61),*rev_2W=PP(62),*rev_2b=PP(63),
    *rev_sW=PP(64),*rev_sb=PP(65),
    *csW=PP(66),*csb=PP(67),*ceW=PP(68),*ceb=PP(69),
    *kW_=PP(70),*qW_=PP(71),
    *pe1W=PP(72),*pe1b=PP(73),*pe2W=PP(74),*pe2b=PP(75),
    *px1W=PP(76),*px_g=PP(77),*px_be=PP(78),*px2W=PP(79),
    *wm1W=PP(80),*wm1b=PP(81),*wm_g=PP(82),*wm_be=PP(83),
    *wmTW=PP(84),*wmTb=PP(85),
    *dx_1W=PP(86),*dx_1b=PP(87),*dx_2W=PP(88),*dx_2b=PP(89),
    *dx_sW=PP(90),*dx_sb=PP(91);
  (void)wmTb; // softmax over k is shift-invariant -> bias cancels

  const size_t BC_N  = (size_t)BATCH * 128 * NPTS;     // 1,048,576
  const size_t B64N  = (size_t)BATCH * 64 * NPTS;      //   524,288
  const size_t G8_64 = (size_t)BATCH * 64 * NPTS * 8;  // 4,194,304
  const size_t G8_128= (size_t)BATCH * 128 * NPTS * 8;
  const size_t G16_64= (size_t)BATCH * 64 * NPTS * 16;
  const size_t G16_256=(size_t)BATCH * 256 * NPTS * 16;
  const size_t PCD_SZ = (size_t)BATCH * 3 * NUP;       //    49,152
  const size_t FEAT_SZ= (size_t)BATCH * 128 * NUP;     // 2,097,152

  float* out_pcd  = (float*)d_out;
  float* out_feat = out_pcd + PCD_SZ;
  float* out_cons = out_feat + FEAT_SZ;

  // ---- persistent workspace ----
  Arena P{(char*)d_ws, 0};
  float* posf  = P.f((size_t)BATCH * NPTS * 3);
  float* seedt = P.f((size_t)BATCH * 3 * MSEED);
  int* idx16 = P.i((size_t)BATCH * NPTS * KINT);
  int* idx8s = P.i((size_t)BATCH * NPTS * KSUB);
  int* idx8p = P.i((size_t)BATCH * NPTS * KSUB);
  float* prop_fea = P.f(BC_N);
  float* fea2     = P.f(BC_N);
  float* Qb       = P.f(BC_N);
  float* valueB   = P.f(BC_N);
  float* keyB     = P.f(B64N);
  float* queryB   = P.f(B64N);
  float* valueS   = P.f(B64N);
  float* proxyB   = P.f(B64N);
  float* aggup    = P.f((size_t)BATCH * 64 * NUP);
  const size_t scratch0 = P.off;

  dim3 gN(cdiv(NPTS, 256), BATCH);

  // ---- phase 0: layouts + KNN ----
  pos_to_points_kernel<<<gN, 256, 0, stream>>>(pos, posf, NPTS);
  points_to_ch_kernel<<<dim3(cdiv(MSEED, 256), BATCH), 256, 0, stream>>>(seed, seedt, MSEED);
  knn_kernel<KINT><<<gN, 256, 0, stream>>>(posf, posf, idx16, NPTS, NPTS);
  knn_kernel<KSUB><<<gN, 256, 0, stream>>>(posf, posf, idx8s, NPTS, NPTS);
  knn_kernel<KSUB><<<gN, 256, 0, stream>>>(posf, seed, idx8p, NPTS, MSEED);

  // ---- phase A: fea_prop -> prop_fea ----
  {
    Arena T{(char*)d_ws, scratch0};
    float* vs  = T.f((size_t)BATCH * 64 * MSEED);
    float* kf  = T.f((size_t)BATCH * 64 * MSEED);
    float* vf  = T.f((size_t)BATCH * 64 * MSEED);
    float* qv  = T.f(B64N);
    float* qk  = T.f(G8_64);
    float* vgr = T.f(G8_64);
    float* prl = T.f((size_t)BATCH * 3 * NPTS * 8);
    float* h4  = T.f((size_t)BATCH * 4 * NPTS * 8);
    float* pea = T.f(G8_64);
    float* pe  = T.f(G8_64);
    float* w1  = T.f(G8_128);
    float* w   = T.f(G8_64);
    float* ag  = T.f(B64N);
    gemm(stream, pr_sW, seed_feat, vs, pr_sb, 0, 0, 64, 128, MSEED, 0);
    gemm(stream, pr_kW, vs, kf, pr_kb, 0, 0, 64, 64, MSEED, 0);
    gemm(stream, pr_vW, vs, vf, pr_vb, 0, 0, 64, 64, MSEED, 0);
    gemm(stream, pr_qW, fea, qv, pr_qb, 0, 0, 64, 128, NPTS, 0);
    group(stream, kf, idx8p, qv, qk, 64, MSEED, NPTS, 8, 1);         // q - k
    group(stream, vf, idx8p, 0, vgr, 64, MSEED, NPTS, 8, 0);
    group(stream, seedt, idx8p, pos, prl, 3, MSEED, NPTS, 8, 1);     // pos_rel
    build_h_kernel<<<dim3(cdiv(NPTS * 8, 256), BATCH), 256, 0, stream>>>(prl, h4, NPTS * 8);
    gemm(stream, pr_pm1W, h4, pea, pr_pm1b, pr_pm_g, pr_pm_be, 64, 4, NPTS * 8, 1);
    gemm(stream, pr_pm2W, pea, pe, pr_pm2b, 0, 0, 64, 64, NPTS * 8, 0);
    ew_add(stream, qk, qk, pe, G8_64);
    gemm(stream, pr_am1W, qk, w1, pr_am1b, pr_am_g, pr_am_be, 128, 64, NPTS * 8, 1);
    gemm(stream, pr_am2W, w1, w, pr_am2b, 0, 0, 64, 128, NPTS * 8, 0);
    softmax_rows_kernel<<<cdiv(B64N, 256), 256, 0, stream>>>(w, B64N, 8);
    ew_add(stream, vgr, vgr, pe, G8_64);
    attn_agg_kernel<<<cdiv(B64N, 256), 256, 0, stream>>>(w, vgr, ag, B64N, 8);
    gemm(stream, pr_eW, ag, prop_fea, pr_eb, 0, 0, 128, 64, NPTS, 0);
    ew_add(stream, prop_fea, prop_fea, fea, BC_N);
  }

  // ---- phase B: cross_attn -> fea2 ----
  {
    Arena T{(char*)d_ws, scratch0};
    float* cat1 = T.f((size_t)BATCH * 256 * NPTS);
    float* v0   = T.f(BC_N);
    float* valc = T.f(BC_N);
    float* qc = T.f(B64N); float* kf = T.f(B64N); float* vf = T.f(B64N);
    float* qk = T.f(G8_64); float* vgr = T.f(G8_64);
    float* prl = T.f((size_t)BATCH * 3 * NPTS * 8);
    float* pea = T.f(G8_64); float* pe = T.f(G8_64);
    float* w1 = T.f(G8_128); float* w = T.f(G8_64);
    float* ag = T.f(B64N); float* et = T.f(BC_N);
    copy_ch(stream, cat1, 256, 0, prop_fea, 128, NPTS);
    copy_ch(stream, cat1, 256, 128, fea, 128, NPTS);
    gemm(stream, ca_vm1W, cat1, v0, ca_vm1b, 0, 0, 128, 256, NPTS, 1);
    gemm(stream, ca_vm2W, v0, valc, ca_vm2b, 0, 0, 128, 128, NPTS, 0);
    gemm(stream, ca_qW, prop_fea, qc, 0, 0, 0, 64, 128, NPTS, 0);
    gemm(stream, ca_kW, fea, kf, 0, 0, 0, 64, 128, NPTS, 0);
    gemm(stream, ca_vW, valc, vf, ca_vb, 0, 0, 64, 128, NPTS, 0);
    group(stream, kf, idx8s, qc, qk, 64, NPTS, NPTS, 8, 1);
    group(stream, pos, idx8s, pos, prl, 3, NPTS, NPTS, 8, 1);
    gemm(stream, ca_pm1W, prl, pea, ca_pm1b, ca_pm_g, ca_pm_be, 64, 3, NPTS * 8, 1);
    gemm(stream, ca_pm2W, pea, pe, ca_pm2b, 0, 0, 64, 64, NPTS * 8, 0);
    group(stream, vf, idx8s, 0, vgr, 64, NPTS, NPTS, 8, 0);
    ew_add(stream, vgr, vgr, pe, G8_64);
    ew_add(stream, qk, qk, pe, G8_64);
    gemm(stream, ca_am1W, qk, w1, ca_am1b, ca_am_g, ca_am_be, 128, 64, NPTS * 8, 1);
    gemm(stream, ca_am2W, w1, w, ca_am2b, 0, 0, 64, 128, NPTS * 8, 0);
    softmax_rows_kernel<<<cdiv(B64N, 256), 256, 0, stream>>>(w, B64N, 8);
    attn_agg_kernel<<<cdiv(B64N, 256), 256, 0, stream>>>(w, vgr, ag, B64N, 8);
    gemm(stream, ca_eW, ag, et, ca_eb, 0, 0, 128, 64, NPTS, 0);
    ew_avg(stream, fea2, et, valc, BC_N);
  }

  // ---- phase Q: mlp_1 / maxpool / mlp_2 -> Q ----
  {
    Arena T{(char*)d_ws, scratch0};
    float* t32 = T.f((size_t)BATCH * 32 * NPTS);
    float* f1  = T.f(B64N);
    float* f1m = T.f((size_t)BATCH * 64);
    float* cat = T.f((size_t)BATCH * 256 * NPTS);
    float* mt  = T.f(BC_N);
    gemm(stream, m1aW, pos, t32, m1ab, 0, 0, 32, 3, NPTS, 1);
    gemm(stream, m1bW, t32, f1, m1bb, 0, 0, 64, 32, NPTS, 0);
    rowmax_kernel<<<cdiv(BATCH * 64, 256), 256, 0, stream>>>(f1, f1m, BATCH * 64, NPTS);
    copy_ch(stream, cat, 256, 0, f1, 64, NPTS);
    bcast_ch_kernel<<<dim3(cdiv((size_t)64 * NPTS, 256), BATCH), 256, 0, stream>>>(cat, 256, 64, f1m, 64, NPTS);
    copy_ch(stream, cat, 256, 128, prop_fea, 128, NPTS);
    gemm(stream, m2aW, cat, mt, m2ab, 0, 0, 128, 256, NPTS, 1);
    gemm(stream, m2bW, mt, Qb, m2bb, 0, 0, 128, 128, NPTS, 0);
  }

  // ---- phase C: value/dup/cons + key/query/valueS/proxy ----
  {
    Arena T{(char*)d_ws, scratch0};
    float* cat = T.f((size_t)BATCH * 256 * NPTS);
    float* mh  = T.f(BC_N); float* a = T.f(BC_N); float* b2 = T.f(BC_N);
    copy_ch(stream, cat, 256, 0, Qb, 128, NPTS);
    copy_ch(stream, cat, 256, 128, fea2, 128, NPTS);
    gemm(stream, map_1W, cat, mh, map_1b, 0, 0, 128, 256, NPTS, 1);
    gemm(stream, map_2W, mh, a, map_2b, 0, 0, 128, 128, NPTS, 0);
    gemm(stream, map_sW, cat, b2, map_sb, 0, 0, 128, 256, NPTS, 0);
    ew_avg(stream, valueB, a, b2, BC_N);
    copy_ch(stream, cat, 256, 128, valueB, 128, NPTS); // Q still in ch 0..127
    gemm(stream, rev_1W, cat, mh, rev_1b, 0, 0, 128, 256, NPTS, 1);
    gemm(stream, rev_2W, mh, a, rev_2b, 0, 0, 128, 128, NPTS, 0);
    gemm(stream, rev_sW, cat, b2, rev_sb, 0, 0, 128, 256, NPTS, 0);
    ew_avg(stream, mh, a, b2, BC_N); // dup
    ew_sub_kernel<<<cdiv(BC_N, 256), 256, 0, stream>>>(out_cons, valueB, mh, BC_N);
    gemm(stream, kW_, valueB, keyB, 0, 0, 0, 64, 128, NPTS, 0);
    gemm(stream, qW_, Qb, queryB, 0, 0, 0, 64, 128, NPTS, 0);
    gemm(stream, csW, valueB, valueS, csb, 0, 0, 64, 128, NPTS, 0);
    float* pa = T.f((size_t)BATCH * 32 * NPTS);
    float* pb = T.f(B64N); float* pu = T.f(B64N);
    gemm(stream, pe1W, pos, pa, pe1b, 0, 0, 32, 3, NPTS, 1);
    gemm(stream, pe2W, pa, pb, pe2b, 0, 0, 64, 32, NPTS, 0);
    gemm(stream, upW, fea2, pu, upb, 0, 0, 64, 128, NPTS, 0);
    ew_add(stream, proxyB, pu, pb, B64N);
  }

  // ---- phase D: k=16 upsampling attention -> aggup ----
  {
    Arena T{(char*)d_ws, scratch0};
    float* qk2 = T.f(G16_64);
    float* pr1 = T.f(G16_64);
    float* big = T.f(G16_256);   // shared by px-mid and wm-mid
    float* pr2 = T.f(G16_64);
    float* s0  = T.f(G16_64); float* s1 = T.f(G16_64);
    float* vg  = T.f(G16_64);
    float* W0  = T.f(64 * 256); float* W1 = T.f(64 * 256);
    group(stream, keyB, idx16, queryB, qk2, 64, NPTS, NPTS, 16, 1);    // query - key
    group(stream, proxyB, idx16, proxyB, pr1, 64, NPTS, NPTS, 16, 2);  // g - proxy
    gemm(stream, px1W, pr1, big, 0, px_g, px_be, 256, 64, NPTS * 16, 1);
    gemm(stream, px2W, big, pr2, 0, 0, 0, 64, 256, NPTS * 16, 0);
    ew_add(stream, qk2, qk2, pr2, G16_64);
    gemm(stream, wm1W, qk2, big, wm1b, wm_g, wm_be, 256, 64, NPTS * 16, 1);
    pack_wmT_kernel<<<cdiv(64 * 256, 256), 256, 0, stream>>>(wmTW, W0, W1);
    gemm(stream, W0, big, s0, 0, 0, 0, 64, 256, NPTS * 16, 0);
    gemm(stream, W1, big, s1, 0, 0, 0, 64, 256, NPTS * 16, 0);
    group(stream, valueS, idx16, valueS, vg, 64, NPTS, NPTS, 16, 2);   // relative values
    upsample_attn_kernel<<<cdiv(B64N, 256), 256, 0, stream>>>(s0, s1, vg, aggup, B64N, NPTS);
  }

  // ---- phase E: child features + displacement ----
  {
    Arena T{(char*)d_ws, scratch0};
    float* y   = T.f(FEAT_SZ);
    float* rid = T.f(FEAT_SZ);
    gemm(stream, ceW, aggup, y, ceb, 0, 0, 128, 64, NUP, 0);
    repeat2_kernel<<<cdiv(BC_N, 256), 256, 0, stream>>>(valueB, rid, BATCH * 128, NPTS);
    ew_add(stream, out_feat, y, rid, FEAT_SZ);
    float* fcr = T.f(FEAT_SZ);
    ew_relu_kernel<<<cdiv(FEAT_SZ, 256), 256, 0, stream>>>(fcr, out_feat, FEAT_SZ);
    float* dxh = T.f((size_t)BATCH * 32 * NUP);
    float* a3 = T.f(PCD_SZ); float* b3 = T.f(PCD_SZ);
    float* mo = T.f(PCD_SZ); float* rp = T.f(PCD_SZ);
    gemm(stream, dx_1W, fcr, dxh, dx_1b, 0, 0, 32, 128, NUP, 1);
    gemm(stream, dx_2W, dxh, a3, dx_2b, 0, 0, 3, 32, NUP, 0);
    gemm(stream, dx_sW, fcr, b3, dx_sb, 0, 0, 3, 128, NUP, 0);
    ew_avg(stream, mo, a3, b3, PCD_SZ);
    repeat2_kernel<<<cdiv((size_t)BATCH * 3 * NPTS, 256), 256, 0, stream>>>(pos, rp, BATCH * 3, NPTS);
    pcd_kernel<<<cdiv(PCD_SZ, 256), 256, 0, stream>>>(out_pcd, rp, mo, PCD_SZ);
  }
#undef PP
}
